// RSSMCore_23905787970089
// MI455X (gfx1250) — matrix operand order repaired and checked by reading, not compiled, not run on hardware
//
#include <hip/hip_runtime.h>
#include <hip/hip_bf16.h>
#include <cstdint>
#include <cstddef>

// Problem dims (compile-time, from reference)
#define TT 64
#define BB 512
#define EE 1024
#define AA 16
#define DD 512
#define SS 32
#define HH 512
#define MIN_STD 0.1f
#define LOG2E 1.4426950408889634f
#define LN2   0.6931471805599453f

typedef __attribute__((ext_vector_type(16))) __bf16 v16bf;
typedef __attribute__((ext_vector_type(8)))  __bf16 bf16x8;
typedef __attribute__((ext_vector_type(8)))  float  v8f;
typedef __attribute__((ext_vector_type(4)))  float  f32x4;

static __device__ __forceinline__ v8f wmma_bf16(v16bf a, v16bf b, v8f c) {
  // D = A(16x32 bf16) * B(32x16 bf16) + C(f32)
  return __builtin_amdgcn_wmma_f32_16x16x32_bf16(false, a, false, b, (short)0, c,
                                                 false, false);
}

// ---- fast transcendentals (hardware TRANS ops, co-execute with matrix pipe)
static __device__ __forceinline__ float fexp(float x) {
  return __builtin_amdgcn_exp2f(x * LOG2E);
}
static __device__ __forceinline__ float frcp(float x) {
  return __builtin_amdgcn_rcpf(x);
}
static __device__ __forceinline__ float eluf(float x) {
  return x > 0.f ? x : fexp(x) - 1.f;
}
static __device__ __forceinline__ float sigmf(float x) {
  return frcp(1.f + fexp(-x));
}
static __device__ __forceinline__ float tanh_fast(float x) {
  float a = fabsf(x);
  float t = fexp(-2.f * a);
  float r = (1.f - t) * frcp(1.f + t);
  return copysignf(r, x);
}
static __device__ __forceinline__ float softplusf(float x) {
  return x > 20.f ? x : __builtin_amdgcn_logf(1.f + fexp(x)) * LN2;
}

// ---- operand builders -------------------------------------------------------
// A operand, fp32 source (small per-step inputs only); optional row scale.
static __device__ __forceinline__ v16bf a_tile_f32(const float* X, int ldx, int m,
                                                   int k0, int hi, float scale) {
  const float* p = X + (size_t)m * ldx + k0 + hi * 8;
  f32x4 q0 = *(const f32x4*)(p);
  f32x4 q1 = *(const f32x4*)(p + 4);
  f32x4 q2 = *(const f32x4*)(p + 16);
  f32x4 q3 = *(const f32x4*)(p + 20);
  v16bf a;
#pragma unroll
  for (int i = 0; i < 4; ++i) {
    a[i]      = (__bf16)(q0[i] * scale);
    a[4 + i]  = (__bf16)(q1[i] * scale);
    a[8 + i]  = (__bf16)(q2[i] * scale);
    a[12 + i] = (__bf16)(q3[i] * scale);
  }
  return a;
}

// A operand, bf16 source (global or LDS; no converts)
static __device__ __forceinline__ v16bf a_tile_bf(const __bf16* X, int ldx, int m,
                                                  int k0, int hi) {
  const __bf16* p = X + (size_t)m * ldx + k0 + hi * 8;
  bf16x8 q0 = *(const bf16x8*)(p);
  bf16x8 q1 = *(const bf16x8*)(p + 16);
  v16bf a;
#pragma unroll
  for (int i = 0; i < 8; ++i) { a[i] = q0[i]; a[8 + i] = q1[i]; }
  return a;
}

// A operand, bf16 source with row-zeroing (reset mask) via select
static __device__ __forceinline__ v16bf a_tile_bf_z(const __bf16* X, int ldx, int m,
                                                    int k0, int hi, bool zero) {
  const __bf16* p = X + (size_t)m * ldx + k0 + hi * 8;
  bf16x8 q0 = *(const bf16x8*)(p);
  bf16x8 q1 = *(const bf16x8*)(p + 16);
  bf16x8 zv = {};
  if (zero) { q0 = zv; q1 = zv; }
  v16bf a;
#pragma unroll
  for (int i = 0; i < 8; ++i) { a[i] = q0[i]; a[8 + i] = q1[i]; }
  return a;
}

// B operand: weights W[nrows, K] row-major bf16; B[k][n] = W[n][k].
static __device__ __forceinline__ v16bf b_tile(const __bf16* W, int ldw, int n,
                                               int k0, int hi) {
  const __bf16* p = W + (size_t)n * ldw + k0 + hi * 8;
  bf16x8 q0 = *(const bf16x8*)(p);
  bf16x8 q1 = *(const bf16x8*)(p + 16);
  v16bf b;
#pragma unroll
  for (int i = 0; i < 8; ++i) { b[i] = q0[i]; b[8 + i] = q1[i]; }
  return b;
}

// ---- utility kernels --------------------------------------------------------
__global__ void __launch_bounds__(256) k_cvt_bf16(const float* __restrict__ s,
                                                  __bf16* __restrict__ d, int nn) {
  int i = blockIdx.x * 256 + threadIdx.x;
  if (i < nn) d[i] = (__bf16)s[i];
}
__global__ void __launch_bounds__(256) k_copy_f32(const float* __restrict__ s,
                                                  float* __restrict__ d, int nn) {
  int i = blockIdx.x * 256 + threadIdx.x;
  if (i < nn) d[i] = s[i];
}

// ---- big parallel GEMM: Y[M,512] = X[M,K] @ W^T (+bias, opt elu) -----------
// Block tile 128x64: 8 waves stacked in M, each wave 16 rows x 64 cols
// (4 accumulators), so each A load feeds 4 WMMAs.
template <int KDIM, bool ELU, bool OUT_BF16>
__global__ void __launch_bounds__(256)
k_gemm_wide(const __bf16* __restrict__ X, const __bf16* __restrict__ W,
            const float* __restrict__ bias, float* __restrict__ yf,
            __bf16* __restrict__ ybf) {
  const int lane = threadIdx.x & 31;
  const int wave = threadIdx.x >> 5;
  const int row0 = blockIdx.x * 128 + wave * 16;
  const int col0 = blockIdx.y * 64;
  const int hi = lane >> 4;
  const int ln = lane & 15;
  const int m = row0 + ln;
  v8f a0 = {}, a1 = {}, a2 = {}, a3 = {};
  for (int kc = 0; kc < KDIM; kc += 32) {
    v16bf a = a_tile_bf(X, KDIM, m, kc, hi);
    a0 = wmma_bf16(a, b_tile(W, KDIM, col0 + ln,      kc, hi), a0);
    a1 = wmma_bf16(a, b_tile(W, KDIM, col0 + 16 + ln, kc, hi), a1);
    a2 = wmma_bf16(a, b_tile(W, KDIM, col0 + 32 + ln, kc, hi), a2);
    a3 = wmma_bf16(a, b_tile(W, KDIM, col0 + 48 + ln, kc, hi), a3);
  }
  v8f* accs[4] = {&a0, &a1, &a2, &a3};
#pragma unroll
  for (int j = 0; j < 4; ++j) {
    const int col = col0 + j * 16 + ln;
    const float b = bias[col];
#pragma unroll
    for (int r = 0; r < 8; ++r) {
      const int row = row0 + r + 8 * hi;
      float v = (*accs[j])[r] + b;
      if (ELU) v = eluf(v);
      if (OUT_BF16) ybf[(size_t)row * 512 + col] = (__bf16)v;
      else          yf[(size_t)row * 512 + col] = v;
    }
  }
}

// ---- one 16x16 tile of za = elu(z_masked@Wz^T + a@Wa^T + bz) -> LDS --------
static __device__ __forceinline__ void za_tile(
    const float* __restrict__ zst, const float* __restrict__ act_t,
    const unsigned char* __restrict__ reset_t, const __bf16* __restrict__ wz,
    const __bf16* __restrict__ wa, const float* __restrict__ bz,
    int gr0, int gc0, int hi, int ln, __bf16* lds, int lr0) {
  const int m = gr0 + ln;
  const int n = gc0 + ln;
  const float mk = reset_t[m] ? 0.f : 1.f;
  v8f acc = {};
  // chunk 0: K = S = 32 (stochastic state, reset-masked)
  {
    v16bf a = a_tile_f32(zst, SS, m, 0, hi, mk);
    acc = wmma_bf16(a, b_tile(wz, SS, n, 0, hi), acc);
  }
  // chunk 1: K = A = 16, zero-padded to 32
  {
    const float* ap = act_t + (size_t)m * AA + hi * 8;
    f32x4 q0 = *(const f32x4*)(ap);
    f32x4 q1 = *(const f32x4*)(ap + 4);
    v16bf a;
#pragma unroll
    for (int i = 0; i < 4; ++i) { a[i] = (__bf16)q0[i]; a[4 + i] = (__bf16)q1[i]; }
#pragma unroll
    for (int i = 8; i < 16; ++i) a[i] = (__bf16)0.f;
    v16bf b;
    {
      bf16x8 wq = *(const bf16x8*)(wa + (size_t)n * AA + hi * 8);
#pragma unroll
      for (int i = 0; i < 8; ++i) { b[i] = wq[i]; b[8 + i] = (__bf16)0.f; }
    }
    acc = wmma_bf16(a, b, acc);
  }
  const float bias = bz[n];
#pragma unroll
  for (int r = 0; r < 8; ++r) {
    lds[(size_t)(lr0 + r + 8 * hi) * HH + n] = (__bf16)eluf(acc[r] + bias);
  }
}

// ---- fused: za (-> LDS) + GRU gates -> h_new   [B, D] ----------------------
// Grid (B/64, D/32); block computes za[64 rows x 512] in LDS, then gates.
__global__ void __launch_bounds__(256)
k_za_gates(const float* __restrict__ zst, const float* __restrict__ act_t,
           const unsigned char* __restrict__ reset_t,
           const __bf16* __restrict__ wz, const __bf16* __restrict__ wa,
           const float* __restrict__ bz, const __bf16* __restrict__ hprev_bf,
           const float* __restrict__ hprev, const __bf16* __restrict__ wih,
           const __bf16* __restrict__ whh, const float* __restrict__ bih,
           const float* __restrict__ bhh, float* __restrict__ hnext,
           __bf16* __restrict__ sh_bf_t, float* __restrict__ feat_t) {
  __shared__ __bf16 zs[64 * HH];  // 64 KB
  __builtin_prefetch(wih, 0, 3);
  __builtin_prefetch(whh, 0, 3);
  const int lane = threadIdx.x & 31;
  const int wave = threadIdx.x >> 5;
  const int hi = lane >> 4;
  const int ln = lane & 15;
  const int brow0 = blockIdx.x * 64;

  // Phase 1: za for this block's 64 rows, all 512 cols (128 tiles, 16/wave)
  for (int i = 0; i < 16; ++i) {
    int tj = wave * 16 + i;
    int tr = tj & 3;
    int tc = tj >> 2;
    za_tile(zst, act_t, reset_t, wz, wa, bz, brow0 + tr * 16, tc * 16, hi, ln, zs,
            tr * 16);
  }
  __syncthreads();

  // Phase 2: gate GEMMs, A(za) from LDS
  const int row0 = brow0 + (wave & 3) * 16;
  const int col0 = blockIdx.y * 32 + (wave >> 2) * 16;
  const int m = row0 + ln;
  const int ml = m - brow0;
  const int n = col0 + ln;
  const bool rz = reset_t[m] != 0;
  v8f axr = {}, axu = {}, axn = {}, ahr = {}, ahu = {}, ahn = {};
  for (int kc = 0; kc < HH; kc += 32) {
    v16bf aza = a_tile_bf(zs, HH, ml, kc, hi);
    v16bf ah  = a_tile_bf_z(hprev_bf, DD, m, kc, hi, rz);
    axr = wmma_bf16(aza, b_tile(wih, HH, n,          kc, hi), axr);
    axu = wmma_bf16(aza, b_tile(wih, HH, n + DD,     kc, hi), axu);
    axn = wmma_bf16(aza, b_tile(wih, HH, n + 2 * DD, kc, hi), axn);
    ahr = wmma_bf16(ah,  b_tile(whh, DD, n,          kc, hi), ahr);
    ahu = wmma_bf16(ah,  b_tile(whh, DD, n + DD,     kc, hi), ahu);
    ahn = wmma_bf16(ah,  b_tile(whh, DD, n + 2 * DD, kc, hi), ahn);
  }
  const int col = col0 + ln;
  const float bxr = bih[col], bxu = bih[col + DD], bxn = bih[col + 2 * DD];
  const float bhr = bhh[col], bhu = bhh[col + DD], bhn = bhh[col + 2 * DD];
#pragma unroll
  for (int r = 0; r < 8; ++r) {
    int row = row0 + r + 8 * hi;
    float rg = sigmf(axr[r] + bxr + ahr[r] + bhr);
    float ug = sigmf(axu[r] + bxu + ahu[r] + bhu);
    float ng = tanh_fast(axn[r] + bxn + rg * (ahn[r] + bhn));
    float hm = hprev[(size_t)row * DD + col] * (reset_t[row] ? 0.f : 1.f);
    float hv = (1.f - ug) * ng + ug * hm;
    hnext[(size_t)row * DD + col] = hv;
    sh_bf_t[(size_t)row * DD + col] = (__bf16)hv;
    feat_t[(size_t)row * (DD + SS) + col] = hv;
  }
}

// ---- fused: post_in (-> LDS) + posterior head + rsample --------------------
// Grid (B/32); block owns 32 rows. post_in for those rows is computed exactly
// once (no redundancy), staged in LDS, then head GEMM + softplus + sample.
__global__ void __launch_bounds__(256)
k_postin_head(const __bf16* __restrict__ sh_bf_t, const __bf16* __restrict__ wph,
              const float* __restrict__ epre_t, const __bf16* __restrict__ wpo,
              const float* __restrict__ bpo, const float* __restrict__ noise_t,
              float* __restrict__ out_t, float* __restrict__ zst_out,
              float* __restrict__ feat_t) {
  __shared__ __bf16 pi[32 * HH];  // 32 KB
  __shared__ float ms[32][65];
  __builtin_prefetch(wph, 0, 3);
  const int lane = threadIdx.x & 31;
  const int wave = threadIdx.x >> 5;
  const int hi = lane >> 4;
  const int ln = lane & 15;
  const int brow0 = blockIdx.x * 32;

  // Phase 1: post_in[32 x 512] (64 tiles, 8/wave)
  for (int i = 0; i < 8; ++i) {
    int tj = wave * 8 + i;
    int tr = tj & 1;
    int tc = tj >> 1;
    const int gr0 = brow0 + tr * 16;
    const int gc0 = tc * 16;
    const int m = gr0 + ln;
    const int n = gc0 + ln;
    v8f acc = {};
    for (int kc = 0; kc < DD; kc += 32) {
      acc = wmma_bf16(a_tile_bf(sh_bf_t, DD, m, kc, hi),
                      b_tile(wph, DD, n, kc, hi), acc);
    }
#pragma unroll
    for (int r = 0; r < 8; ++r) {
      int row = gr0 + r + 8 * hi;
      float v = acc[r] + epre_t[(size_t)row * HH + n];
      pi[(size_t)(tr * 16 + r + 8 * hi) * HH + n] = (__bf16)eluf(v);
    }
  }
  __syncthreads();

  // Phase 2: head GEMM [32 x 64] from LDS; 8 waves = 2(M) x 4(N)
  const int rl0 = (wave & 1) * 16;
  const int c0 = (wave >> 1) * 16;
  const int mlh = rl0 + ln;
  const int nh = c0 + ln;
  v8f acc = {};
  for (int kc = 0; kc < HH; kc += 32) {
    acc = wmma_bf16(a_tile_bf(pi, HH, mlh, kc, hi), b_tile(wpo, HH, nh, kc, hi),
                    acc);
  }
  const float bias = bpo[nh];
#pragma unroll
  for (int r = 0; r < 8; ++r) {
    int rl = rl0 + r + 8 * hi;
    float v = acc[r] + bias;
    if (nh >= SS) v = softplusf(v) + MIN_STD;  // std half
    out_t[(size_t)(brow0 + rl) * (2 * SS) + nh] = v;
    ms[rl][nh] = v;
  }
  __syncthreads();

  // Phase 3: sample = mean + std*noise (32 rows x 32 dims, 4/thread)
#pragma unroll
  for (int it = 0; it < 4; ++it) {
    int idx = threadIdx.x + it * 256;
    int rl = idx >> 5;
    int s = idx & 31;
    int row = brow0 + rl;
    float smp = ms[rl][s] + ms[rl][SS + s] * noise_t[(size_t)row * SS + s];
    zst_out[(size_t)row * SS + s] = smp;
    feat_t[(size_t)row * (DD + SS) + DD + s] = smp;
  }
}

// ---- prior head (no sampling): [T*B, 64] -----------------------------------
__global__ void __launch_bounds__(256)
k_head_plain(const __bf16* __restrict__ xin, const __bf16* __restrict__ wo,
             const float* __restrict__ bo, float* __restrict__ out) {
  const int lane = threadIdx.x & 31;
  const int wave = threadIdx.x >> 5;
  const int row0 = blockIdx.x * 64 + (wave & 3) * 16;
  const int col0 = blockIdx.y * 32 + (wave >> 2) * 16;
  const int hi = lane >> 4;
  const int ln = lane & 15;
  const int m = row0 + ln;
  const int n = col0 + ln;
  v8f acc = {};
  for (int kc = 0; kc < HH; kc += 32) {
    acc = wmma_bf16(a_tile_bf(xin, HH, m, kc, hi), b_tile(wo, HH, n, kc, hi), acc);
  }
  const float bias = bo[n];
#pragma unroll
  for (int r = 0; r < 8; ++r) {
    int row = row0 + r + 8 * hi;
    float v = acc[r] + bias;
    if (n >= SS) v = softplusf(v) + MIN_STD;
    out[(size_t)row * (2 * SS) + n] = v;
  }
}

// ---------------------------------------------------------------------------
extern "C" void kernel_launch(void* const* d_in, const int* in_sizes, int n_in,
                              void* d_out, int out_size, void* d_ws, size_t ws_size,
                              hipStream_t stream) {
  (void)in_sizes; (void)n_in; (void)out_size; (void)ws_size;
  const float* embed  = (const float*)d_in[0];
  const float* action = (const float*)d_in[1];
  const unsigned char* reset = (const unsigned char*)d_in[2];
  /* d_in[3] glob_state: unused (as in reference) */
  const float* noise = (const float*)d_in[4];
  const float* h0 = (const float*)d_in[5];
  const float* z0 = (const float*)d_in[6];
  const float* Wz  = (const float*)d_in[7];
  const float* bz  = (const float*)d_in[8];
  const float* Wa  = (const float*)d_in[9];
  const float* Wih = (const float*)d_in[10];
  const float* bih = (const float*)d_in[11];
  const float* Whh = (const float*)d_in[12];
  const float* bhh = (const float*)d_in[13];
  const float* Wph = (const float*)d_in[14];
  const float* bph = (const float*)d_in[15];
  const float* Wpe = (const float*)d_in[16];
  const float* Wpo = (const float*)d_in[17];
  const float* bpo = (const float*)d_in[18];
  const float* Wprh = (const float*)d_in[19];
  const float* bprh = (const float*)d_in[20];
  const float* Wpr  = (const float*)d_in[21];
  const float* bpr  = (const float*)d_in[22];

  float* out = (float*)d_out;
  float* out_priors = out;
  float* out_posts  = out_priors + (size_t)TT * BB * 2 * SS;
  float* out_feat   = out_posts  + (size_t)TT * BB * 2 * SS;
  float* out_hT     = out_feat   + (size_t)TT * BB * (DD + SS);
  float* out_zT     = out_hT     + (size_t)BB * DD;

  // workspace carve (256B aligned)
  uint8_t* wp = (uint8_t*)d_ws;
  auto alloc = [&](size_t bytes) -> void* {
    void* p = (void*)wp;
    wp += (bytes + 255) & ~(size_t)255;
    return p;
  };
  float*  epre       = (float*)alloc(sizeof(float) * (size_t)TT * BB * HH);
  __bf16* embed_bf   = (__bf16*)alloc(2ull * TT * BB * EE);
  __bf16* sh_bf      = (__bf16*)alloc(2ull * TT * BB * DD);
  __bf16* priorin_bf = (__bf16*)alloc(2ull * TT * BB * HH);
  __bf16* wz_bf  = (__bf16*)alloc(2ull * HH * SS);
  __bf16* wa_bf  = (__bf16*)alloc(2ull * HH * AA);
  __bf16* wih_bf = (__bf16*)alloc(2ull * 3 * DD * HH);
  __bf16* whh_bf = (__bf16*)alloc(2ull * 3 * DD * DD);
  __bf16* wph_bf = (__bf16*)alloc(2ull * HH * DD);
  __bf16* wpe_bf = (__bf16*)alloc(2ull * HH * EE);
  __bf16* wpo_bf = (__bf16*)alloc(2ull * 2 * SS * HH);
  __bf16* wprh_bf = (__bf16*)alloc(2ull * HH * DD);
  __bf16* wpr_bf  = (__bf16*)alloc(2ull * 2 * SS * HH);
  __bf16* h0_bf = (__bf16*)alloc(2ull * BB * DD);
  float* hbuf0 = (float*)alloc(sizeof(float) * (size_t)BB * DD);
  float* hbuf1 = (float*)alloc(sizeof(float) * (size_t)BB * DD);
  float* zst   = (float*)alloc(sizeof(float) * (size_t)BB * SS);

  auto cvt = [&](const float* s, __bf16* d, int nn) {
    k_cvt_bf16<<<(nn + 255) / 256, 256, 0, stream>>>(s, d, nn);
  };
  cvt(Wz, wz_bf, HH * SS);
  cvt(Wa, wa_bf, HH * AA);
  cvt(Wih, wih_bf, 3 * DD * HH);
  cvt(Whh, whh_bf, 3 * DD * DD);
  cvt(Wph, wph_bf, HH * DD);
  cvt(Wpe, wpe_bf, HH * EE);
  cvt(Wpo, wpo_bf, 2 * SS * HH);
  cvt(Wprh, wprh_bf, HH * DD);
  cvt(Wpr, wpr_bf, 2 * SS * HH);
  cvt(embed, embed_bf, TT * BB * EE);
  cvt(h0, h0_bf, BB * DD);

  k_copy_f32<<<(BB * DD + 255) / 256, 256, 0, stream>>>(h0, hbuf0, BB * DD);
  k_copy_f32<<<(BB * SS + 255) / 256, 256, 0, stream>>>(z0, zst, BB * SS);

  // Parallel precompute: Epre = embed @ Wpe^T + bph over all T (no elu, f32 out)
  k_gemm_wide<EE, false, false>
      <<<dim3(TT * BB / 128, HH / 64), 256, 0, stream>>>(embed_bf, wpe_bf, bph,
                                                         epre, (__bf16*)nullptr);

  // Sequential scan: 2 kernels per step
  const float* hprev = hbuf0;
  float* hnext = hbuf1;
  for (int t = 0; t < TT; ++t) {
    const unsigned char* rst = reset + (size_t)t * BB;
    const float* act_t = action + (size_t)t * BB * AA;
    float* feat_t = out_feat + (size_t)t * BB * (DD + SS);
    __bf16* sh_t = sh_bf + (size_t)t * BB * DD;
    const __bf16* hprev_bf = (t == 0) ? h0_bf : (sh_bf + (size_t)(t - 1) * BB * DD);

    k_za_gates<<<dim3(BB / 64, DD / 32), 256, 0, stream>>>(
        zst, act_t, rst, wz_bf, wa_bf, bz, hprev_bf, hprev, wih_bf, whh_bf, bih,
        bhh, hnext, sh_t, feat_t);
    k_postin_head<<<BB / 32, 256, 0, stream>>>(
        sh_t, wph_bf, epre + (size_t)t * BB * HH, wpo_bf, bpo,
        noise + (size_t)t * BB * SS, out_posts + (size_t)t * BB * 2 * SS, zst,
        feat_t);

    const float* tp = hnext; hnext = (float*)hprev; hprev = tp;  // ping-pong
  }

  // Final carry
  k_copy_f32<<<(BB * DD + 255) / 256, 256, 0, stream>>>(hprev, out_hT, BB * DD);
  k_copy_f32<<<(BB * SS + 255) / 256, 256, 0, stream>>>(zst, out_zT, BB * SS);

  // Prior network over all T (parallel): elu GEMM (bf16 out) then head
  k_gemm_wide<DD, true, true>
      <<<dim3(TT * BB / 128, HH / 64), 256, 0, stream>>>(
          sh_bf, wprh_bf, bprh, (float*)nullptr, priorin_bf);
  k_head_plain<<<dim3(TT * BB / 64, 2 * SS / 32), 256, 0, stream>>>(
      priorin_bf, wpr_bf, bpr, out_priors);
}
